// Rel_Attention_56169582297329
// MI455X (gfx1250) — matrix operand order repaired and checked by reading, not compiled, not run on hardware
//
#include <hip/hip_runtime.h>

typedef __bf16 bf16_t;
typedef __attribute__((ext_vector_type(4)))  __bf16 v4bf;
typedef __attribute__((ext_vector_type(8)))  __bf16 v8bf;
typedef __attribute__((ext_vector_type(16))) __bf16 v16bf;
typedef __attribute__((ext_vector_type(4)))  float  v4f;
typedef __attribute__((ext_vector_type(8)))  float  v8f;

#define DIM   512
#define HEADS 16
#define NTOK  64
#define HD    32
#define QK_SCALE 0.17677669529663687f  // 32^-0.5

// LDS strides in elements (all multiples of 8 halfs -> 16B-aligned vector access)
#define LXS  520   // x tile stride (halfs)
#define LWS  520   // weight tile stride (halfs)
#define LQK  36    // q/k f32 stride (floats)
#define LQKH 40    // q/k bf16 stride (halfs)
#define LVT  72    // v^T bf16 stride (halfs)
#define LAT  68    // attn f32 stride (floats)
#define LATH 72    // attn probs bf16 stride (halfs)

// LDS byte offsets
#define OFF_LX    0                     // 2*64*520*2   = 133120
#define OFF_LW    133120                // 96*520*2     =  99840
#define OFF_LQK   232960                // 128*36*4     =  18432
#define OFF_LQKH  251392                // 128*40*2     =  10240
#define OFF_LVT   261632                // 32*72*2      =   4608
#define OFF_LAT   266240                // 64*68*4      =  17408
#define OFF_LATH  283648                // 64*72*2      =   9216
#define OFF_LBIAS 292864                // 225*4 -> pad 912
#define OFF_LQB   293776                // 96*4
#define SMEM_BYTES 294160

__device__ __forceinline__ v16bf load_frag(const bf16_t* p, int delta) {
  union { v16bf v; v8bf h[2]; } u;
  u.h[0] = *(const v8bf*)p;
  u.h[1] = *(const v8bf*)(p + delta);
  return u.v;
}

__global__ void cvt_w_bf16(const float* __restrict__ w, bf16_t* __restrict__ o, int n) {
  int i = blockIdx.x * blockDim.x + threadIdx.x;
  int stride = gridDim.x * blockDim.x;
  for (; i < n; i += stride) o[i] = (bf16_t)w[i];
}

__global__ __launch_bounds__(256) void win_attn_fused(
    const float* __restrict__ x, const bf16_t* __restrict__ wbf,
    const float* __restrict__ qkv_b, const float* __restrict__ bias_table,
    float* __restrict__ out)
{
  extern __shared__ __align__(16) char smem[];
  bf16_t* lx    = (bf16_t*)(smem + OFF_LX);    // [2][64][LXS] bf16 x tiles
  bf16_t* lw    = (bf16_t*)(smem + OFF_LW);    // [96][LWS] head weight slice (rows = out channel)
  float*  lqk   = (float*) (smem + OFF_LQK);   // [128][LQK] q rows 0-63, k rows 64-127 (f32)
  bf16_t* lqkh  = (bf16_t*)(smem + OFF_LQKH);  // [128][LQKH] normalized q/k (bf16)
  bf16_t* lvT   = (bf16_t*)(smem + OFF_LVT);   // [32][LVT] v transposed: [dh][token]
  float*  lat   = (float*) (smem + OFF_LAT);   // [64][LAT] attn scores (f32)
  bf16_t* lath  = (bf16_t*)(smem + OFF_LATH);  // [64][LATH] softmax probs (bf16)
  float*  lbias = (float*) (smem + OFF_LBIAS); // [225] rel-pos bias column for head h
  float*  lqb   = (float*) (smem + OFF_LQB);   // [96] qkv bias slice for head h

  const int tid  = threadIdx.x;
  const int wave = tid >> 5;
  const int lane = tid & 31;
  const int ln   = lane & 15;
  const int hi   = lane >> 4;
  const int b0   = blockIdx.x * 2;

  // ---- Phase 0: stage x for 2 windows as bf16 ----
  {
    const float* xg = x + (size_t)b0 * NTOK * DIM;
    for (int idx = tid; idx < 2 * NTOK * DIM / 4; idx += 256) {
      int e = idx * 4;
      v4f f = *(const v4f*)(xg + e);
      int wdw = e >> 15;            // / (64*512)
      int rem = e & 32767;
      int row = rem >> 9;
      int col = rem & 511;
      v4bf hh;
      hh.x = (bf16_t)f.x; hh.y = (bf16_t)f.y; hh.z = (bf16_t)f.z; hh.w = (bf16_t)f.w;
      *(v4bf*)(lx + (size_t)(wdw * NTOK + row) * LXS + col) = hh;
    }
  }
  __syncthreads();

  for (int h = 0; h < HEADS; ++h) {
    // ---- stage head weight slice (q/k/v rows), rel-bias column, qkv bias ----
    for (int t = tid; t < 96 * 64; t += 256) {
      int j   = t >> 6;             // local out-channel 0..95
      int seg = t & 63;             // 8-half segment along K
      int gr  = (j >> 5) * DIM + h * HD + (j & 31);  // global qkv_w row
      *(v8bf*)(lw + (size_t)j * LWS + seg * 8) =
          *(const v8bf*)(wbf + (size_t)gr * DIM + seg * 8);
    }
    if (tid < 225) lbias[tid] = bias_table[tid * HEADS + h];
    if (tid < 96)  lqb[tid]   = qkv_b[(tid >> 5) * DIM + h * HD + (tid & 31)];
    __syncthreads();

    // ---- prefetch next head's weight slice into cache while we compute ----
    // 96 rows x 1024B each = 768 cachelines of 128B -> 3 global_prefetch per thread
    if (h + 1 < HEADS) {
      for (int t = tid; t < 96 * 8; t += 256) {
        int j   = t >> 3;                                   // row 0..95
        int seg = t & 7;                                    // 128B segment
        int gr  = (j >> 5) * DIM + (h + 1) * HD + (j & 31);
        __builtin_prefetch(wbf + (size_t)gr * DIM + seg * 64, 0, 3);
      }
    }

    for (int w = 0; w < 2; ++w) {
      const bf16_t* lxw = lx + (size_t)w * NTOK * LXS;

      // ---- Phase 2: QKV GEMM: [64 x 512] x [512 x 96] -> q,k,v ----
      // 4 M-tiles x 6 N-tiles; wave owns one M-tile and 3 N-tiles (A reuse, ILP=3)
      {
        const int mt  = wave & 3;
        const int ntb = (wave >> 2) * 3;
        v8f acc[3] = {v8f{}, v8f{}, v8f{}};
        for (int ks = 0; ks < 16; ++ks) {
          v16bf a = load_frag(lxw + (size_t)(mt * 16 + ln) * LXS + ks * 32 + hi * 8, 16);
          #pragma unroll
          for (int j = 0; j < 3; ++j) {
            v16bf b = load_frag(lw + (size_t)((ntb + j) * 16 + ln) * LWS + ks * 32 + hi * 8, 16);
            acc[j] = __builtin_amdgcn_wmma_f32_16x16x32_bf16(
                false, a, false, b, (short)0, acc[j], false, false);
          }
        }
        #pragma unroll
        for (int j = 0; j < 3; ++j) {
          int nt  = ntb + j;
          int col = nt * 16 + ln;
          float bias = lqb[col];
          #pragma unroll
          for (int r = 0; r < 8; ++r) {
            float val = acc[j][r] + bias;
            int tok = mt * 16 + hi * 8 + r;
            if (nt < 2)       lqk[tok * LQK + col] = val;                         // q
            else if (nt < 4)  lqk[(64 + tok) * LQK + (col - 32)] = val;           // k
            else              lvT[(col - 64) * LVT + tok] = (bf16_t)val;          // v^T
          }
        }
      }
      __syncthreads();

      // ---- Phase 3: cosine-normalize q (w/ scale) and k rows ----
      if (tid < 128) {
        float ss = 0.f;
        #pragma unroll
        for (int c = 0; c < HD; ++c) { float v = lqk[tid * LQK + c]; ss += v * v; }
        float inv = rsqrtf(ss);
        if (tid < 64) inv *= QK_SCALE;
        #pragma unroll
        for (int c = 0; c < HD; ++c)
          lqkh[tid * LQKH + c] = (bf16_t)(lqk[tid * LQK + c] * inv);
      }
      __syncthreads();

      // ---- Phase 4: attn = qn . kn^T + rel_bias ; 4x4 tiles, 2 per wave ----
      {
        const int mt  = wave & 3;
        const int ntb = (wave >> 2) * 2;
        v16bf a = load_frag(lqkh + (size_t)(mt * 16 + ln) * LQKH + hi * 8, 16);
        #pragma unroll
        for (int j = 0; j < 2; ++j) {
          int nt = ntb + j;
          v16bf b = load_frag(lqkh + (size_t)(64 + nt * 16 + ln) * LQKH + hi * 8, 16);
          v8f acc = {};
          acc = __builtin_amdgcn_wmma_f32_16x16x32_bf16(
              false, a, false, b, (short)0, acc, false, false);
          int n = nt * 16 + ln;
          int nh = n >> 3, nw = n & 7;
          #pragma unroll
          for (int r = 0; r < 8; ++r) {
            int m = mt * 16 + hi * 8 + r;
            int mh = m >> 3, mw = m & 7;
            int idx = (mh - nh + 7) * 15 + (mw - nw + 7);
            lat[m * LAT + n] = acc[r] + lbias[idx];
          }
        }
      }
      __syncthreads();

      // ---- Phase 5: softmax rows -> bf16 probs ----
      if (tid < 64) {
        float mx = -1e30f;
        #pragma unroll
        for (int c = 0; c < 64; ++c) mx = fmaxf(mx, lat[tid * LAT + c]);
        float s = 0.f;
        #pragma unroll
        for (int c = 0; c < 64; ++c) s += __expf(lat[tid * LAT + c] - mx);
        float invs = 1.0f / s;
        #pragma unroll
        for (int c = 0; c < 64; ++c)
          lath[tid * LATH + c] = (bf16_t)(__expf(lat[tid * LAT + c] - mx) * invs);
      }
      __syncthreads();

      // ---- Phase 6: out = P . V ; 4x2 tiles, 1 per wave ----
      {
        const int mt = wave & 3;
        const int nt = wave >> 2;
        v8f acc = {};
        #pragma unroll
        for (int ks = 0; ks < 2; ++ks) {
          v16bf a = load_frag(lath + (size_t)(mt * 16 + ln) * LATH + ks * 32 + hi * 8, 16);
          v16bf b = load_frag(lvT + (size_t)(nt * 16 + ln) * LVT + ks * 32 + hi * 8, 16);
          acc = __builtin_amdgcn_wmma_f32_16x16x32_bf16(
              false, a, false, b, (short)0, acc, false, false);
        }
        int dh = nt * 16 + ln;
        float* og = out + (size_t)(b0 + w) * NTOK * DIM + h * HD + dh;
        #pragma unroll
        for (int r = 0; r < 8; ++r) {
          int tok = mt * 16 + hi * 8 + r;
          og[(size_t)tok * DIM] = acc[r];
        }
      }
      __syncthreads();
    } // windows
  } // heads
}

extern "C" void kernel_launch(void* const* d_in, const int* in_sizes, int n_in,
                              void* d_out, int out_size, void* d_ws, size_t ws_size,
                              hipStream_t stream) {
  (void)in_sizes; (void)n_in; (void)out_size; (void)ws_size;
  const float* x          = (const float*)d_in[0];
  const float* qkv_w      = (const float*)d_in[1];
  const float* qkv_b      = (const float*)d_in[2];
  const float* bias_table = (const float*)d_in[3];
  float* out = (float*)d_out;
  bf16_t* wbf = (bf16_t*)d_ws;   // 3*512*512 bf16 = 1.5 MB scratch

  const int nw = 3 * DIM * DIM;
  hipLaunchKernelGGL(cvt_w_bf16, dim3(512), dim3(256), 0, stream, qkv_w, wbf, nw);
  hipLaunchKernelGGL(win_attn_fused, dim3(4096 / 2), dim3(256), SMEM_BYTES, stream,
                     x, wbf, qkv_b, bias_table, out);
}
